// SiGAT_77627238908562
// MI455X (gfx1250) — compile-verified, matrix operand order checked by reading, not run on hardware
//
#include <hip/hip_runtime.h>
#include <math.h>

// ---------------- problem constants (from reference) ----------------
#define N_NODES 30000
#define N_REL   38
#define N_EDGES 200000
#define HID     64
#define ETOT    (N_EDGES + N_NODES)   // edges + self loops
#define NEG_SLOPE 0.2f

typedef __attribute__((ext_vector_type(16))) _Float16 v16h;
typedef __attribute__((ext_vector_type(8)))  float    v8f;

// ---------------- WMMA 16x16x32 f16 GEMM:  C[M,64] (+)= A[M,64] @ B[64,64] ----------------
// B panel (16 KB f32) staged in LDS once per block, shared by 4 waves.
// One wave per 16-row M tile, full N=64 (4 n-tiles), K=64 in two 32-wide steps.
// mode 0: C = A@B (+ bias[n] if bias != nullptr)     mode 1: C += A@B
__global__ __launch_bounds__(128)
void gemm64_wmma(const float* __restrict__ A, const float* __restrict__ B,
                 float* __restrict__ C, const float* __restrict__ bias,
                 int M, int accumulate) {
    __shared__ float Bs[64 * 64];
    {   // cooperative stage: 1024 float4, 8 per thread
        const float4* Bv = (const float4*)B;
        float4*       Sv = (float4*)Bs;
#pragma unroll
        for (int t = 0; t < 8; ++t)
            Sv[threadIdx.x + 128 * t] = Bv[threadIdx.x + 128 * t];
    }
    __syncthreads();

    const int lane   = threadIdx.x & 31;
    const int wave   = threadIdx.x >> 5;
    const int mTile  = blockIdx.x * 4 + wave;
    const int mTiles = M >> 4;                  // M is a multiple of 16 (30000)
    if (mTile >= mTiles) return;                // wave-uniform: EXEC all-ones inside
    const int m0   = mTile << 4;
    const int row  = lane & 15;                 // M row (A) / N col (B,C,D)
    const int half = lane >> 4;                 // lane-half K/M split per ISA layout

    // B fragments from LDS: 2 k-steps x 4 n-tiles.  B[k][n] row-major; n = row.
    v16h bf[2][4];
#pragma unroll
    for (int s = 0; s < 2; ++s)
#pragma unroll
        for (int j = 0; j < 4; ++j) {
            const int n = j * 16 + row;
#pragma unroll
            for (int i = 0; i < 8; ++i) {
                const int kb = (i < 4 ? 2 * i : 16 + 2 * (i - 4)) + 8 * half + 32 * s;
                bf[s][j][2 * i + 0] = (_Float16)Bs[(kb + 0) * 64 + n];
                bf[s][j][2 * i + 1] = (_Float16)Bs[(kb + 1) * 64 + n];
            }
        }

    // Accumulator init: bias broadcast / read-modify-write for fused accumulation.
    v8f acc[4];
#pragma unroll
    for (int j = 0; j < 4; ++j) {
        const int n = j * 16 + row;
#pragma unroll
        for (int i = 0; i < 8; ++i) {
            const int r = m0 + i + 8 * half;    // C/D: VGPR i -> row m0+i (+8 for lane half)
            float c0 = 0.0f;
            if (accumulate)      c0 = C[r * 64 + n];
            else if (bias)       c0 = bias[n];
            acc[j][i] = c0;
        }
    }

    // A fragments + WMMA chain.
#pragma unroll
    for (int s = 0; s < 2; ++s) {
        const float* Arow = A + (m0 + row) * 64 + 32 * s;
        __builtin_prefetch(Arow + 4 * 16 * 64, 0, 0);   // next block's A rows
        v16h af;
#pragma unroll
        for (int i = 0; i < 8; ++i) {
            const int kb = (i < 4 ? 2 * i : 16 + 2 * (i - 4)) + 8 * half;
            af[2 * i + 0] = (_Float16)Arow[kb + 0];
            af[2 * i + 1] = (_Float16)Arow[kb + 1];
        }
#pragma unroll
        for (int j = 0; j < 4; ++j)
            acc[j] = __builtin_amdgcn_wmma_f32_16x16x32_f16(
                false, af, false, bf[s][j], (short)0, acc[j], false, false);
    }

    // Store D.
#pragma unroll
    for (int j = 0; j < 4; ++j) {
        const int n = j * 16 + row;
#pragma unroll
        for (int i = 0; i < 8; ++i)
            C[(m0 + i + 8 * half) * 64 + n] = acc[j][i];
    }
}

// ---------------- per-node attention scores ----------------
__global__ void scores_kernel(const float* __restrict__ h,
                              const float* __restrict__ att_src,
                              const float* __restrict__ att_dst,
                              float* __restrict__ s_src, float* __restrict__ s_dst) {
    int n = blockIdx.x * blockDim.x + threadIdx.x;
    if (n >= N_NODES) return;
    const float4* hv = (const float4*)(h + (size_t)n * 64);
    const float4* as = (const float4*)att_src;
    const float4* ad = (const float4*)att_dst;
    float ss = 0.f, sd = 0.f;
#pragma unroll
    for (int i = 0; i < 16; ++i) {
        float4 x = hv[i], a = as[i], b = ad[i];
        ss += x.x * a.x + x.y * a.y + x.z * a.z + x.w * a.w;
        sd += x.x * b.x + x.y * b.y + x.z * b.z + x.w * b.w;
    }
    s_src[n] = ss;
    s_dst[n] = sd;
}

// ---------------- CSR build: counts -> scan -> bucket fill ----------------
__global__ void zero_ints(int* __restrict__ p, int n) {
    int i = blockIdx.x * blockDim.x + threadIdx.x;
    if (i < n) p[i] = 0;
}

__global__ void count_edges(const int* __restrict__ dst, int* __restrict__ counts) {
    int i = blockIdx.x * blockDim.x + threadIdx.x;
    if (i >= ETOT) return;
    int d = (i < N_EDGES) ? dst[i] : (i - N_EDGES);   // self loop past E
    atomicAdd(counts + d, 1);
}

// single-workgroup chunked Hillis-Steele exclusive scan over N_NODES counts
__global__ __launch_bounds__(1024)
void scan_offsets(const int* __restrict__ counts,
                  int* __restrict__ offsets, int* __restrict__ cursor) {
    __shared__ int lds[1024];
    __shared__ int carry;
    if (threadIdx.x == 0) carry = 0;
    __syncthreads();
    for (int base = 0; base < N_NODES; base += 1024) {
        int idx = base + (int)threadIdx.x;
        int v   = (idx < N_NODES) ? counts[idx] : 0;
        lds[threadIdx.x] = v;
        __syncthreads();
#pragma unroll
        for (int off = 1; off < 1024; off <<= 1) {
            int t = (threadIdx.x >= (unsigned)off) ? lds[threadIdx.x - off] : 0;
            __syncthreads();
            lds[threadIdx.x] += t;
            __syncthreads();
        }
        int excl = lds[threadIdx.x] - v;
        if (idx < N_NODES) {
            int o = carry + excl;
            offsets[idx] = o;
            cursor[idx]  = o;
        }
        __syncthreads();                 // all carry reads done
        if (threadIdx.x == 1023) carry += lds[1023];
        __syncthreads();
    }
    if (threadIdx.x == 0) offsets[N_NODES] = carry;
}

__global__ void fill_csr(const int* __restrict__ src, const int* __restrict__ dst,
                         int* __restrict__ cursor, int* __restrict__ srcs) {
    int i = blockIdx.x * blockDim.x + threadIdx.x;
    if (i >= ETOT) return;
    int s = (i < N_EDGES) ? src[i] : (i - N_EDGES);
    int d = (i < N_EDGES) ? dst[i] : (i - N_EDGES);
    int pos = atomicAdd(cursor + d, 1);
    srcs[pos] = s;
}

// ---------------- GAT per-destination softmax + aggregation (gather, no atomics) ----------------
// One wave32 per destination node; features 64 = 2 per lane.
__global__ __launch_bounds__(256)
void gat_node(const int* __restrict__ offsets, const int* __restrict__ srcs,
              const float* __restrict__ s_src, const float* __restrict__ s_dst,
              const float* __restrict__ h, const float* __restrict__ bias,
              float* __restrict__ out) {
    int d    = (blockIdx.x * blockDim.x + threadIdx.x) >> 5;
    int lane = threadIdx.x & 31;
    if (d >= N_NODES) return;
    int beg = offsets[d], end = offsets[d + 1];      // >= 1 edge (self loop)
    float sd = s_dst[d];

    // pass 1: segment max (lane-parallel + shuffle reduce)
    float lmax = -3.0e38f;
    for (int j = beg + lane; j < end; j += 32) {
        float e = s_src[srcs[j]] + sd;
        e = (e > 0.f) ? e : NEG_SLOPE * e;
        lmax = fmaxf(lmax, e);
    }
#pragma unroll
    for (int o = 16; o; o >>= 1) lmax = fmaxf(lmax, __shfl_xor(lmax, o, 32));

    // pass 2: softmax denominator
    float lsum = 0.f;
    for (int j = beg + lane; j < end; j += 32) {
        float e = s_src[srcs[j]] + sd;
        e = (e > 0.f) ? e : NEG_SLOPE * e;
        lsum += __expf(e - lmax);
    }
#pragma unroll
    for (int o = 16; o; o >>= 1) lsum += __shfl_xor(lsum, o, 32);
    float inv = 1.f / lsum;

    // pass 3: weighted gather-accumulate of h[src] rows (coalesced 256 B per edge)
    float o0 = 0.f, o1 = 0.f;
    for (int j = beg; j < end; ++j) {
        int   s = srcs[j];                            // wave-uniform broadcast load
        float e = s_src[s] + sd;
        e = (e > 0.f) ? e : NEG_SLOPE * e;
        float alpha = __expf(e - lmax) * inv;
        const float* hs = h + (size_t)s * 64;
        o0 = fmaf(alpha, hs[lane],      o0);
        o1 = fmaf(alpha, hs[lane + 32], o1);
    }
    out[(size_t)d * 64 + lane]      = o0 + bias[lane];
    out[(size_t)d * 64 + lane + 32] = o1 + bias[lane + 32];
}

__global__ void tanh_inplace(float* __restrict__ x, int n) {
    int i = blockIdx.x * blockDim.x + threadIdx.x;
    if (i < n) x[i] = tanhf(x[i]);
}

// ---------------- host driver ----------------
extern "C" void kernel_launch(void* const* d_in, const int* in_sizes, int n_in,
                              void* d_out, int out_size, void* d_ws, size_t ws_size,
                              hipStream_t stream) {
    const float* emb     = (const float*)d_in[0];   // [N,64]
    const float* W       = (const float*)d_in[1];   // [R,64,64]
    const float* att_src = (const float*)d_in[2];   // [R,64]
    const float* att_dst = (const float*)d_in[3];   // [R,64]
    const float* bias    = (const float*)d_in[4];   // [R,64]
    const float* W1      = (const float*)d_in[5];   // [(R+1)*64, 64]
    const float* b1      = (const float*)d_in[6];   // [64]
    const float* W2      = (const float*)d_in[7];   // [64,64]
    const float* b2      = (const float*)d_in[8];   // [64]
    const int*   edges   = (const int*)d_in[9];     // [R,2,E]
    // d_in[10] = nodes (arange) -> identity gather, unused.

    // Workspace layout (~25 MB, L2-resident on MI455X's 192 MB L2).
    float* ws  = (float*)d_ws;
    float* acc     = ws;  ws += (size_t)N_NODES * 64;  // fused MLP layer-1 accumulator
    float* h       = ws;  ws += (size_t)N_NODES * 64;  // per-relation h = emb @ W_r
    float* outr    = ws;  ws += (size_t)N_NODES * 64;  // per-relation GAT output
    float* s_src   = ws;  ws += N_NODES;
    float* s_dst   = ws;  ws += N_NODES;
    int*   counts  = (int*)ws;  ws += N_NODES;
    int*   cursor  = (int*)ws;  ws += N_NODES;
    int*   offsets = (int*)ws;  ws += (N_NODES + 1);
    int*   srcs    = (int*)ws;  ws += ETOT;

    const int gemmGrid = ((N_NODES / 16) + 3) / 4;           // 4 waves/block, 1 M-tile/wave
    const int nodeGrid = (N_NODES + 255) / 256;
    const int elemGrid = (N_NODES * 64 + 255) / 256;
    const int edgeGrid = (ETOT + 255) / 256;
    const int waveGrid = (int)(((size_t)N_NODES * 32 + 255) / 256);

    // acc = emb @ W1[0:64,:] + b1   (first block of the fused MLP layer 1)
    gemm64_wmma<<<gemmGrid, 128, 0, stream>>>(emb, W1, acc, b1, N_NODES, 0);

    for (int r = 0; r < N_REL; ++r) {
        const float* Wr   = W + (size_t)r * 64 * 64;
        const int*   srcp = edges + (size_t)r * 2 * N_EDGES;
        const int*   dstp = srcp + N_EDGES;

        gemm64_wmma<<<gemmGrid, 128, 0, stream>>>(emb, Wr, h, nullptr, N_NODES, 0);
        scores_kernel<<<nodeGrid, 256, 0, stream>>>(h, att_src + r * 64, att_dst + r * 64,
                                                    s_src, s_dst);
        // CSR build for this relation
        zero_ints<<<nodeGrid, 256, 0, stream>>>(counts, N_NODES);
        count_edges<<<edgeGrid, 256, 0, stream>>>(dstp, counts);
        scan_offsets<<<1, 1024, 0, stream>>>(counts, offsets, cursor);
        fill_csr<<<edgeGrid, 256, 0, stream>>>(srcp, dstp, cursor, srcs);
        // softmax + aggregation, gather-based (no float atomics)
        gat_node<<<waveGrid, 256, 0, stream>>>(offsets, srcs, s_src, s_dst, h,
                                               bias + r * 64, outr);
        // fused MLP layer 1: acc += out_r @ W1[(r+1)*64 : (r+2)*64, :]
        gemm64_wmma<<<gemmGrid, 128, 0, stream>>>(outr, W1 + (size_t)(r + 1) * 64 * 64,
                                                  acc, nullptr, N_NODES, 1);
    }

    tanh_inplace<<<elemGrid, 256, 0, stream>>>(acc, N_NODES * 64);
    gemm64_wmma<<<gemmGrid, 128, 0, stream>>>(acc, W2, (float*)d_out, b2, N_NODES, 0);
}
